// GraphFilter_36155034697800
// MI455X (gfx1250) — compile-verified, matrix operand order checked by reading, not run on hardware
//
#include <hip/hip_runtime.h>
#include <hip/hip_bf16.h>
#include <math.h>

// ---------------------------------------------------------------------------
// GraphFilter propagate for MI455X (gfx1250), fused, A never materialized.
//   out = (1/3) X + (2/3) * ( D^-1/2 A D^-1/2 X + D^-1 X )
//   A   = thresh( Fn Fn^T ),  diag(A)=0,  D = diag(rowsum(A)+1)
// bf16 WMMA (v_wmma_f32_16x16x32_bf16) for both N^2 K=512 GEMMs, f32 accum.
// ---------------------------------------------------------------------------

typedef __bf16 bf16;
typedef bf16  v8bf  __attribute__((ext_vector_type(8)));
typedef bf16  v16bf __attribute__((ext_vector_type(16)));
typedef float v8f   __attribute__((ext_vector_type(8)));

#define EPS_THRESH 1e-10f
#define NORM_EPS   1e-12f
#define REG        (2.0f / 3.0f)

// Load a v16bf A/B-operand fragment for wmma_f32_16x16x32_bf16.
// Per ISA 16-bit A-matrix layout: lanes 0-15 hold K = {k..k+7, k+16..k+23},
// lanes 16-31 hold K = {k+8..k+15, k+24..k+31}. Caller passes
// k = kb + (lane>=16 ? 8 : 0); elements 0..7 come from rowptr[k..k+7],
// elements 8..15 from rowptr[k+16..k+23]. Both are 16-byte loads.
__device__ __forceinline__ v16bf load_frag(const bf16* __restrict__ rowptr, int k) {
    v8bf lo = *(const v8bf*)(rowptr + k);
    v8bf hi = *(const v8bf*)(rowptr + k + 16);
    return __builtin_shufflevector(lo, hi,
        0, 1, 2, 3, 4, 5, 6, 7, 8, 9, 10, 11, 12, 13, 14, 15);
}

__device__ __forceinline__ v8f zero_v8f() {
    v8f z = {0.f, 0.f, 0.f, 0.f, 0.f, 0.f, 0.f, 0.f};
    return z;
}

// ---------------------------------------------------------------------------
// Kernel 1: row L2-normalize -> bf16 Fbf[N][F]; also XT[F][N] = bf16(X^T).
// One block (256 threads) per row; F = 512 -> 2 elements per thread.
// ---------------------------------------------------------------------------
__global__ __launch_bounds__(256)
void gf_prep_kernel(const float* __restrict__ X,
                    bf16* __restrict__ Fbf,
                    bf16* __restrict__ XT,
                    int N, int F) {
    const int row = blockIdx.x;
    const int tid = threadIdx.x;
    __shared__ float red[256];

    const float x0 = X[(size_t)row * F + tid];
    const float x1 = X[(size_t)row * F + tid + 256];
    red[tid] = x0 * x0 + x1 * x1;
    __syncthreads();
    for (int s = 128; s > 0; s >>= 1) {
        if (tid < s) red[tid] += red[tid + s];
        __syncthreads();
    }
    const float nrm = fmaxf(sqrtf(red[0]), NORM_EPS);
    const float inv = 1.0f / nrm;

    Fbf[(size_t)row * F + tid]       = (bf16)(x0 * inv);
    Fbf[(size_t)row * F + tid + 256] = (bf16)(x1 * inv);
    XT[(size_t)tid * N + row]         = (bf16)x0;
    XT[(size_t)(tid + 256) * N + row] = (bf16)x1;
}

// ---------------------------------------------------------------------------
// Kernel 2: rowsum[i] = sum_j thresh(Fn_i . Fn_j) (diag zeroed) + 1.0
// 128 threads = 4 waves; wave w owns rows [blk*64 + w*16, +16).
// Each wave sweeps all N columns with 16x16x32 bf16 WMMA tiles.
// ---------------------------------------------------------------------------
__global__ __launch_bounds__(128)
void gf_rowsum_kernel(const bf16* __restrict__ Fbf,
                      float* __restrict__ rowsum,
                      int N, int F) {
    const int wave = threadIdx.x >> 5;
    const int lane = threadIdx.x & 31;
    const int half = lane >> 4;      // 0: lanes 0-15, 1: lanes 16-31
    const int l16  = lane & 15;
    const int i0   = blockIdx.x * 64 + wave * 16;

    const bf16* __restrict__ Arow = Fbf + (size_t)(i0 + l16) * F;

    float racc[8] = {0.f, 0.f, 0.f, 0.f, 0.f, 0.f, 0.f, 0.f};

    for (int j0 = 0; j0 < N; j0 += 16) {
        const bf16* __restrict__ Brow = Fbf + (size_t)(j0 + l16) * F;
        v8f sacc = zero_v8f();
        for (int kb = 0; kb < F; kb += 32) {
            const int ko = kb + half * 8;
            v16bf a = load_frag(Arow, ko);
            v16bf b = load_frag(Brow, ko);
            sacc = __builtin_amdgcn_wmma_f32_16x16x32_bf16(
                false, a, false, b, (short)0, sacc, false, false);
        }
        // C layout: lane holds col N=l16, rows M = v + 8*half.
        const int gcol = j0 + l16;
        for (int v = 0; v < 8; ++v) {
            const int grow = i0 + v + 8 * half;
            float s = sacc[v];
            s = (s < EPS_THRESH) ? 0.f : s;   // threshold
            s = (grow == gcol) ? 0.f : s;     // remove self-loop
            racc[v] += s;
        }
    }

    __shared__ float red[4][32][8];
    for (int v = 0; v < 8; ++v) red[wave][lane][v] = racc[v];
    __syncthreads();

    if (lane < 16) {
        const int r  = lane;           // local row 0..15
        const int hh = (r < 8) ? 0 : 1;
        const int v  = r & 7;
        float s = 0.f;
        for (int c = 0; c < 16; ++c) s += red[wave][hh * 16 + c][v];
        rowsum[i0 + r] = s + 1.0f;     // diag of (A + I)
    }
}

// ---------------------------------------------------------------------------
// Kernel 3: fused propagate. WG = 128 threads (4 waves), owns 16 output rows
// (i0..i0+15) x all F=512 columns. Per 64-wide j-chunk:
//   - wave w computes 16x16 S tile for cols j0+16w (16 WMMAs, K=512),
//     thresholds, scales by d_i^-1/2 d_j^-1/2, zeroes diag, stores bf16
//     A_norm tile to LDS As[16][64];
//   - all waves consume As as WMMA A-operand vs XT B-fragments (16 WMMAs)
//     accumulating the wave's 16x128 f32 output slab.
// Epilogue: out = (1/3)X + (2/3)(acc + X/rowsum).
// ---------------------------------------------------------------------------
__global__ __launch_bounds__(128)
void gf_propagate_kernel(const float* __restrict__ X,
                         const bf16* __restrict__ Fbf,
                         const bf16* __restrict__ XT,
                         const float* __restrict__ rowsum,
                         float* __restrict__ out,
                         int N, int F) {
    const int wave = threadIdx.x >> 5;
    const int lane = threadIdx.x & 31;
    const int half = lane >> 4;
    const int l16  = lane & 15;
    const int i0   = blockIdx.x * 16;   // 16 output rows per WG
    const int fb   = wave * 128;        // wave's feature-column base (8 n-tiles)

    __shared__ __align__(16) bf16 As[16][64];   // A_norm staging tile (2 KB)

    v8f acc[8];
    for (int nt = 0; nt < 8; ++nt) acc[nt] = zero_v8f();

    // Per-lane row scaling constants: rows m = v + 8*half of this WG's tile.
    float dinv_i[8], inv_rs[8];
    for (int v = 0; v < 8; ++v) {
        const float rs = rowsum[i0 + v + 8 * half];
        dinv_i[v] = rsqrtf(rs);
        inv_rs[v] = 1.0f / rs;
    }

    const bf16* __restrict__ Arow = Fbf + (size_t)(i0 + l16) * F;

    for (int j0 = 0; j0 < N; j0 += 64) {
        // ---- GEMM1: this wave's 16x16 similarity tile at cols jb..jb+15 ----
        const int jb = j0 + wave * 16;
        const bf16* __restrict__ Brow = Fbf + (size_t)(jb + l16) * F;
        v8f sacc = zero_v8f();
        for (int kb = 0; kb < F; kb += 32) {
            const int ko = kb + half * 8;
            v16bf a = load_frag(Arow, ko);
            v16bf b = load_frag(Brow, ko);
            sacc = __builtin_amdgcn_wmma_f32_16x16x32_bf16(
                false, a, false, b, (short)0, sacc, false, false);
        }
        // threshold + sym-normalize + kill diag, stage bf16 into LDS
        const int   gcol  = jb + l16;
        const float dinvj = rsqrtf(rowsum[gcol]);
        for (int v = 0; v < 8; ++v) {
            const int m    = v + 8 * half;
            const int grow = i0 + m;
            float s = sacc[v];
            s = (s < EPS_THRESH) ? 0.f : s;
            s = (grow == gcol) ? 0.f : s;
            s *= dinv_i[v] * dinvj;
            As[m][wave * 16 + l16] = (bf16)s;
        }
        __syncthreads();

        // ---- GEMM2: acc += A_norm(16 x 64) * X(64 x 128-wave-slab) ----
        for (int jj = 0; jj < 64; jj += 32) {
            // A-fragment from LDS: lane row m=l16, K pattern per half.
            const bf16* arow = &As[l16][0];
            v16bf afrag = load_frag(arow, jj + half * 8);
            for (int nt = 0; nt < 8; ++nt) {
                const bf16* __restrict__ xtrow =
                    XT + (size_t)(fb + nt * 16 + l16) * N;
                v16bf bfrag = load_frag(xtrow, j0 + jj + half * 8);
                // speculative prefetch of next chunk's K-slice of this row
                __builtin_prefetch(xtrow + j0 + jj + 64, 0, 1);
                acc[nt] = __builtin_amdgcn_wmma_f32_16x16x32_bf16(
                    false, afrag, false, bfrag, (short)0, acc[nt], false, false);
            }
        }
        __syncthreads();
    }

    // ---- epilogue: out = (1-REG)*X + REG*(acc + X/rowsum) ----
    for (int nt = 0; nt < 8; ++nt) {
        for (int v = 0; v < 8; ++v) {
            const int row = i0 + v + 8 * half;
            const int col = fb + nt * 16 + l16;
            const size_t idx = (size_t)row * F + col;
            const float x = X[idx];
            out[idx] = (1.0f - REG) * x + REG * (acc[nt][v] + x * inv_rs[v]);
        }
    }
}

// ---------------------------------------------------------------------------
// Launch: prep -> rowsum -> propagate (stream-ordered, graph-capture safe).
// Workspace: Fbf (N*F bf16) | XT (F*N bf16) | rowsum (N f32)  ~= 16.8 MB
// ---------------------------------------------------------------------------
extern "C" void kernel_launch(void* const* d_in, const int* in_sizes, int n_in,
                              void* d_out, int out_size, void* d_ws, size_t ws_size,
                              hipStream_t stream) {
    (void)n_in; (void)out_size; (void)ws_size;
    const int F = 512;
    const int N = in_sizes[0] / F;     // 8192

    const float* X  = (const float*)d_in[0];
    float*       out = (float*)d_out;

    bf16*  Fbf    = (bf16*)d_ws;
    bf16*  XT     = Fbf + (size_t)N * F;
    float* rowsum = (float*)(XT + (size_t)F * N);

    gf_prep_kernel<<<N, 256, 0, stream>>>(X, Fbf, XT, N, F);
    gf_rowsum_kernel<<<N / 64, 128, 0, stream>>>(Fbf, rowsum, N, F);
    gf_propagate_kernel<<<N / 16, 128, 0, stream>>>(X, Fbf, XT, rowsum, out, N, F);
}
